// RootRNN_15109694947508
// MI455X (gfx1250) — matrix operand order repaired and checked by reading, not compiled, not run on hardware
//
#include <hip/hip_runtime.h>

// ---------------------------------------------------------------------------
// Persistent-grid teacher-forced RNN scan for MI455X (gfx1250).
//   z_curr = alpha*z + (1-alpha)*x_t
//   z_next = A z_curr + W1 relu(W2 z_curr + h2) + h1
// Weights pinned in LDS as bf16 (4.5MB / 32 WGPs = ~144KB each),
// matvecs via v_wmma_f32_16x16x32_bf16, cross-WGP sync via global spin barrier.
// ---------------------------------------------------------------------------

typedef __attribute__((ext_vector_type(16))) __bf16  v16bf;
typedef __attribute__((ext_vector_type(8)))  float   v8f;
typedef __attribute__((ext_vector_type(4)))  float   v4f;
typedef __attribute__((ext_vector_type(4)))  unsigned int v4u;

#define NWG      32
#define NTHREADS 256
#define T_STEPS  16384
#define M_DIM    512
#define L_DIM    2048
#define W2_ROWS  (L_DIM / NWG)   // 64 rows of W2 per workgroup
#define Z_ROWS   (M_DIM / NWG)   // 16 rows of A/W1 per workgroup

union ABFrag { v4u u[2]; v16bf v; };

__device__ __forceinline__ unsigned short f2bf(float f) {
  union { float f; unsigned u; } x; x.f = f;
  unsigned r = x.u + 0x7fffu + ((x.u >> 16) & 1u);   // round-to-nearest-even
  return (unsigned short)(r >> 16);
}

__device__ __forceinline__ unsigned long long f2bf4(v4f f) {
  return  (unsigned long long)f2bf(f[0])
       | ((unsigned long long)f2bf(f[1]) << 16)
       | ((unsigned long long)f2bf(f[2]) << 32)
       | ((unsigned long long)f2bf(f[3]) << 48);
}

// A-matrix 16x32 bf16 fragment (ISA 7.12.2): lane m<16 holds row m;
// VGPR0-3 = K {base..base+7}, VGPR4-7 = K {base+16..base+23}, base=(lane>=16)*8.
__device__ __forceinline__ v16bf load_a_frag(const unsigned short* W, int ld,
                                             int row0, int k0, int lane) {
  const unsigned short* p = W + (row0 + (lane & 15)) * ld + k0 + ((lane >> 4) << 3);
  ABFrag ab;
  ab.u[0] = *(const v4u*)p;
  ab.u[1] = *(const v4u*)(p + 16);
  return ab.v;
}

// B-matrix 32x16 bf16 fragment with the activation vector replicated across
// all 16 columns: same K order as A: K {base..base+7, base+16..base+23}, base=(lane>=16)*8.
__device__ __forceinline__ v16bf load_b_frag(const unsigned short* x, int k0, int lane) {
  const unsigned short* p = x + k0 + ((lane >> 4) << 3);
  ABFrag ab;
  ab.u[0] = *(const v4u*)p;
  ab.u[1] = *(const v4u*)(p + 16);
  return ab.v;
}

// Device-scope grid barrier (single counter + generation, ws-resident).
__device__ __forceinline__ void grid_barrier(unsigned* cnt, unsigned* gen) {
  __threadfence();          // release all prior global writes
  __syncthreads();
  if (threadIdx.x == 0) {
    unsigned g = __hip_atomic_load(gen, __ATOMIC_ACQUIRE, __HIP_MEMORY_SCOPE_AGENT);
    unsigned arrived = __hip_atomic_fetch_add(cnt, 1u, __ATOMIC_ACQ_REL, __HIP_MEMORY_SCOPE_AGENT);
    if (arrived == NWG - 1) {
      __hip_atomic_store(cnt, 0u, __ATOMIC_RELAXED, __HIP_MEMORY_SCOPE_AGENT);
      __hip_atomic_fetch_add(gen, 1u, __ATOMIC_ACQ_REL, __HIP_MEMORY_SCOPE_AGENT);
    } else {
      while (__hip_atomic_load(gen, __ATOMIC_ACQUIRE, __HIP_MEMORY_SCOPE_AGENT) == g)
        __builtin_amdgcn_s_sleep(1);
    }
  }
  __syncthreads();
  __threadfence();          // acquire: invalidate near caches before re-reading z/r
}

__global__ void __launch_bounds__(NTHREADS, 1)
rnn_scan_kernel(const float* __restrict__ xs_gt, const float* __restrict__ Ag,
                const float* __restrict__ W1g, const float* __restrict__ W2g,
                const float* __restrict__ h1g, const float* __restrict__ h2g,
                const float* __restrict__ alphag, float* __restrict__ out,
                unsigned* ctrl, float* zg, unsigned short* rg) {
  // 149.3 KB static LDS -> one workgroup per WGP, weights resident all run.
  __shared__ __attribute__((aligned(16))) unsigned short sW2[W2_ROWS * M_DIM]; // 64KB
  __shared__ __attribute__((aligned(16))) unsigned short sA [Z_ROWS * M_DIM];  // 16KB
  __shared__ __attribute__((aligned(16))) unsigned short sW1[Z_ROWS * L_DIM];  // 64KB
  __shared__ __attribute__((aligned(16))) unsigned short sZ [M_DIM];           // 1KB
  __shared__ __attribute__((aligned(16))) unsigned short sR [L_DIM];           // 4KB
  __shared__ __attribute__((aligned(16))) float sYr[W2_ROWS];
  __shared__ __attribute__((aligned(16))) float sYz[Z_ROWS];

  const int g    = blockIdx.x;
  const int tid  = threadIdx.x;
  const int lane = tid & 31;
  const int wave = tid >> 5;

  // ---- one-time staging: convert this WG's weight slices fp32 -> bf16 in LDS
  const float* w2base = W2g + (size_t)(W2_ROWS * g) * M_DIM;
  for (int i = tid; i < W2_ROWS * M_DIM; i += NTHREADS) sW2[i] = f2bf(w2base[i]);
  const float* abase = Ag + (size_t)(Z_ROWS * g) * M_DIM;
  for (int i = tid; i < Z_ROWS * M_DIM; i += NTHREADS)  sA[i]  = f2bf(abase[i]);
  const float* w1base = W1g + (size_t)(Z_ROWS * g) * L_DIM;
  for (int i = tid; i < Z_ROWS * L_DIM; i += NTHREADS)  sW1[i] = f2bf(w1base[i]);
  if (tid < Z_ROWS) zg[Z_ROWS * g + tid] = 0.0f;       // z0 = 0

  const float alphav = alphag[0];
  const float beta   = 1.0f - alphav;

  grid_barrier(ctrl, ctrl + 1);

  for (int t = 0; t < T_STEPS; ++t) {
    const float* xrow = xs_gt + (size_t)t * M_DIM;

    // z_curr = alpha*z + (1-alpha)*x_t, built 4-wide (every WG, full vector)
    for (int i = tid * 4; i < M_DIM; i += NTHREADS * 4) {
      v4f z4 = *(const v4f*)(zg + i);
      v4f x4 = *(const v4f*)(xrow + i);
      v4f zc = alphav * z4 + beta * x4;
      *(unsigned long long*)(sZ + i) = f2bf4(zc);
    }
    for (int i = tid; i < W2_ROWS; i += NTHREADS)
      sYr[i] = h2g[W2_ROWS * g + i];
    // warm L2 for next step's x-row while we compute (global_prefetch_b8)
    if (tid < 16 && t + 1 < T_STEPS)
      __builtin_prefetch(xrow + M_DIM + tid * 32, 0, 1);
    __syncthreads();

    // ---- Phase A: r_slice = relu(W2_slice @ z_curr + h2_slice)  (64 rows)
    {
      const int tile  = wave & 3;           // 4 row-tiles of 16
      const int kbase = (wave >> 2) * 256;  // 2 K-halves of 256
      v8f c = {0.f, 0.f, 0.f, 0.f, 0.f, 0.f, 0.f, 0.f};
#pragma unroll
      for (int kc = 0; kc < 8; ++kc) {
        v16bf a = load_a_frag(sW2, M_DIM, tile * 16, kbase + kc * 32, lane);
        v16bf b = load_b_frag(sZ, kbase + kc * 32, lane);
        c = __builtin_amdgcn_wmma_f32_16x16x32_bf16(false, a, false, b,
                                                    (short)0, c, false, false);
      }
      // column 0 lives in lanes 0 (rows 0-7) and 16 (rows 8-15)
      if ((lane & 15) == 0) {
        const int rbase = tile * 16 + ((lane >> 4) << 3);
#pragma unroll
        for (int v = 0; v < 8; ++v) atomicAdd(&sYr[rbase + v], c[v]);
      }
    }
    __syncthreads();
    for (int i = tid; i < W2_ROWS; i += NTHREADS) {
      float rv = sYr[i];
      rg[W2_ROWS * g + i] = f2bf(rv > 0.f ? rv : 0.f);
    }
    grid_barrier(ctrl, ctrl + 1);   // full r now visible device-wide

    // ---- Phase B: z_next_slice = A_slice@z + W1_slice@r + h1_slice (16 rows)
    // gather full r: one b128 per thread (2048 bf16 = 4KB in a single sweep)
    for (int i = tid * 8; i < L_DIM; i += NTHREADS * 8)
      *(v4u*)(sR + i) = *(const v4u*)(rg + i);
    if (tid < Z_ROWS) sYz[tid] = h1g[Z_ROWS * g + tid];
    __syncthreads();
    {
      v8f c = {0.f, 0.f, 0.f, 0.f, 0.f, 0.f, 0.f, 0.f};
      // A contribution: K=512 -> 16 chunks, 2 per wave (branch-free)
#pragma unroll
      for (int j = 0; j < 2; ++j) {
        const int k0 = (wave * 2 + j) * 32;
        v16bf a = load_a_frag(sA, M_DIM, 0, k0, lane);
        v16bf b = load_b_frag(sZ, k0, lane);
        c = __builtin_amdgcn_wmma_f32_16x16x32_bf16(false, a, false, b,
                                                    (short)0, c, false, false);
      }
      // W1 contribution: K=2048 -> 64 chunks, 8 per wave (branch-free)
#pragma unroll
      for (int j = 0; j < 8; ++j) {
        const int k0 = (wave * 8 + j) * 32;
        v16bf a = load_a_frag(sW1, L_DIM, 0, k0, lane);
        v16bf b = load_b_frag(sR, k0, lane);
        c = __builtin_amdgcn_wmma_f32_16x16x32_bf16(false, a, false, b,
                                                    (short)0, c, false, false);
      }
      if ((lane & 15) == 0) {
        const int rbase = (lane >> 4) << 3;
#pragma unroll
        for (int v = 0; v < 8; ++v) atomicAdd(&sYz[rbase + v], c[v]);
      }
    }
    __syncthreads();
    if (tid < Z_ROWS) {
      float zn = sYz[tid];
      zg[Z_ROWS * g + tid] = zn;
      out[(size_t)t * M_DIM + Z_ROWS * g + tid] = zn;
    }
    grid_barrier(ctrl, ctrl + 1);   // z updated before next step reads it
  }
}

__global__ void init_ctrl_kernel(unsigned* ctrl) {
  ctrl[0] = 0u;   // arrival counter
  ctrl[1] = 0u;   // generation
}

extern "C" void kernel_launch(void* const* d_in, const int* in_sizes, int n_in,
                              void* d_out, int out_size, void* d_ws, size_t ws_size,
                              hipStream_t stream) {
  (void)in_sizes; (void)n_in; (void)out_size; (void)ws_size;
  const float* xs = (const float*)d_in[0];
  const float* A  = (const float*)d_in[1];
  const float* W1 = (const float*)d_in[2];
  const float* W2 = (const float*)d_in[3];
  const float* h1 = (const float*)d_in[4];
  const float* h2 = (const float*)d_in[5];
  const float* al = (const float*)d_in[6];

  unsigned*       ctrl = (unsigned*)d_ws;
  float*          zg   = (float*)((char*)d_ws + 256);
  unsigned short* rg   = (unsigned short*)((char*)d_ws + 256 + M_DIM * sizeof(float));

  init_ctrl_kernel<<<1, 1, 0, stream>>>(ctrl);
  rnn_scan_kernel<<<NWG, NTHREADS, 0, stream>>>(xs, A, W1, W2, h1, h2, al,
                                                (float*)d_out, ctrl, zg, rg);
}
